// ARFastWeightSwiGLU_23012434772132
// MI455X (gfx1250) — matrix operand order repaired and checked by reading, not compiled, not run on hardware
//
#include <hip/hip_runtime.h>
#include <hip/hip_bf16.h>
#include <stdint.h>

#define B_     16
#define L_     4608
#define D_     256
#define DH_    512
#define MB_    512
#define UE_    1024
#define FIRST_ 512
#define NC_    4
#define WSCALE 0.01f
#define EPS_   1e-5f

typedef __attribute__((ext_vector_type(16))) __bf16 v16bf;
typedef __attribute__((ext_vector_type(8)))  float  v8f;
typedef unsigned int   u32;
typedef unsigned short u16;

__device__ __forceinline__ u16 f2bf(float x) {
  u32 u = __float_as_uint(x);
  u32 r = ((u >> 16) & 1u) + 0x7FFFu;   // round-to-nearest-even
  return (u16)((u + r) >> 16);
}

// ---- WMMA fragment loaders (CDNA5 ISA 7.12.2 layouts), LDS row stride = 17 dwords (34 halves)
__device__ __forceinline__ v16bf load_fragA(const u32* lds, int row) {
  const int lane = threadIdx.x & 31;
  const u32* p = lds + (row + (lane & 15)) * 17;
  const int kb = (lane >> 4) * 4;       // dword offset of K-pair group
  union { u32 u[8]; v16bf v; } f;
#pragma unroll
  for (int j = 0; j < 4; ++j) { f.u[j] = p[kb + j]; f.u[4 + j] = p[8 + kb + j]; }
  return f.v;
}
__device__ __forceinline__ v16bf load_fragB(const u32* lds, int col) {
  const int lane = threadIdx.x & 31;
  const u32* p = lds + (col + (lane & 15)) * 17;
  const int kb = (lane >> 4) * 4;
  union { u32 u[8]; v16bf v; } f;
#pragma unroll
  for (int j = 0; j < 4; ++j) { f.u[j] = p[kb + j]; f.u[4 + j] = p[8 + kb + j]; }
  return f.v;
}

// ---- Batched bf16 GEMM: C[b][M][N] (+)= A[b][M][K] * B[b][N][K]^T ; K-contiguous operands.
// Block = 128 threads (4 waves), block tile 128x128, wave tile 64x64 (4x4 WMMA 16x16x32).
__global__ void __launch_bounds__(128)
gemm_bf16(const u16* __restrict__ A, int sA,
          const u16* __restrict__ Bm, int sB,
          float* __restrict__ C, int sC,
          int M, int N, int K, int ldc, int acc)
{
  __shared__ u32 lA[128 * 17];
  __shared__ u32 lB[128 * 17];
  const int b  = blockIdx.z;
  const int m0 = blockIdx.y * 128;
  const int n0 = blockIdx.x * 128;
  const u32* Ag = (const u32*)(A + (size_t)b * sA);
  const u32* Bg = (const u32*)(Bm + (size_t)b * sB);
  float* Cg = C + (size_t)b * sC;
  const int tid  = threadIdx.x;
  const int lane = tid & 31;
  const int wave = tid >> 5;
  const int wm = (wave >> 1) * 64;
  const int wn = (wave & 1) * 64;
  const int Kd = K >> 1;                       // dwords per operand row

  v8f c[4][4];
#pragma unroll
  for (int i = 0; i < 4; ++i)
#pragma unroll
    for (int j = 0; j < 4; ++j) c[i][j] = (v8f){};

  for (int k0 = 0; k0 < K; k0 += 32) {
    __syncthreads();
    const int kd = k0 >> 1;
    const int cc = tid & 15;
#pragma unroll
    for (int i = 0; i < 16; ++i) {             // 128 rows x 16 dwords per operand
      int rr = (tid >> 4) + i * 8;
      lA[rr * 17 + cc] = Ag[(size_t)(m0 + rr) * Kd + kd + cc];
      lB[rr * 17 + cc] = Bg[(size_t)(n0 + rr) * Kd + kd + cc];
    }
    if (k0 + 32 < K) {                         // gfx1250 global_prefetch_b8 of next K tile
      __builtin_prefetch(&Ag[(size_t)(m0 + (tid >> 4)) * Kd + kd + 16 + cc], 0, 1);
      __builtin_prefetch(&Bg[(size_t)(n0 + (tid >> 4)) * Kd + kd + 16 + cc], 0, 1);
    }
    __syncthreads();
    v16bf af[4], bf_[4];
#pragma unroll
    for (int i = 0; i < 4; ++i) af[i]  = load_fragA(lA, wm + 16 * i);
#pragma unroll
    for (int j = 0; j < 4; ++j) bf_[j] = load_fragB(lB, wn + 16 * j);
#pragma unroll
    for (int i = 0; i < 4; ++i)
#pragma unroll
      for (int j = 0; j < 4; ++j)
        c[i][j] = __builtin_amdgcn_wmma_f32_16x16x32_bf16(false, af[i], false, bf_[j],
                                                          (short)0, c[i][j], false, false);
  }

  const int mr = (lane >> 4) * 8;
  const int nc = lane & 15;
#pragma unroll
  for (int i = 0; i < 4; ++i)
#pragma unroll
    for (int j = 0; j < 4; ++j) {
      const int mbase = m0 + wm + 16 * i + mr;
      const int nbase = n0 + wn + 16 * j + nc;
#pragma unroll
      for (int r = 0; r < 8; ++r) {
        size_t idx = (size_t)(mbase + r) * ldc + nbase;
        if (acc) Cg[idx] += c[i][j][r];
        else     Cg[idx]  = c[i][j][r];
      }
    }
}

// ---- row L2 norms of initial weights (one block per row)
__global__ void __launch_bounds__(256)
rownorm_kernel(const float* __restrict__ W, float* __restrict__ nrm, int len) {
  __shared__ float red[256];
  const float* row = W + (size_t)blockIdx.x * len;
  float s = 0.f;
  for (int i = threadIdx.x; i < len; i += 256) { float x = row[i]; s += x * x; }
  red[threadIdx.x] = s; __syncthreads();
  for (int off = 128; off > 0; off >>= 1) {
    if (threadIdx.x < off) red[threadIdx.x] += red[threadIdx.x + off];
    __syncthreads();
  }
  if (threadIdx.x == 0) nrm[blockIdx.x] = sqrtf(red[0]);
}

// ---- renormalize rows to stored target norms
__global__ void __launch_bounds__(256)
renorm_kernel(float* __restrict__ W, const float* __restrict__ nrm, int len) {
  __shared__ float red[256];
  __shared__ float fac;
  float* row = W + (size_t)blockIdx.x * len;
  float s = 0.f;
  for (int i = threadIdx.x; i < len; i += 256) { float x = row[i]; s += x * x; }
  red[threadIdx.x] = s; __syncthreads();
  for (int off = 128; off > 0; off >>= 1) {
    if (threadIdx.x < off) red[threadIdx.x] += red[threadIdx.x + off];
    __syncthreads();
  }
  if (threadIdx.x == 0) fac = nrm[blockIdx.x] / (sqrtf(red[0]) + EPS_);
  __syncthreads();
  for (int i = threadIdx.x; i < len; i += 256) row[i] *= fac;
}

__global__ void __launch_bounds__(256)
f32_to_bf16(const float* __restrict__ src, u16* __restrict__ dst, size_t n) {
  size_t i = (size_t)blockIdx.x * 256 + threadIdx.x;
  if (i < n) dst[i] = f2bf(src[i]);
}

// src f32 [B][rows][cols] -> dst bf16 [B][cols][rows]
__global__ void __launch_bounds__(256)
transpose_f32_to_bf16(const float* __restrict__ src, u16* __restrict__ dst, int rows, int cols) {
  size_t idx = (size_t)blockIdx.x * 256 + threadIdx.x;
  size_t n = (size_t)B_ * rows * cols;
  if (idx >= n) return;
  int c = (int)(idx % cols);
  size_t t = idx / cols;
  int r = (int)(t % rows);
  int b = (int)(t / rows);
  dst[(size_t)b * rows * cols + (size_t)c * rows + r] = f2bf(src[idx]);
}

// k/v/lr prep for one update mini-batch
__global__ void __launch_bounds__(256)
prep_kv_kernel(const float* __restrict__ k, const float* __restrict__ v,
               const float* __restrict__ lr0, const float* __restrict__ lr2, int tokOff,
               u16* __restrict__ kb, u16* __restrict__ vb,
               u16* __restrict__ klr0t, u16* __restrict__ klr2t, u16* __restrict__ vbt) {
  size_t idx = (size_t)blockIdx.x * 256 + threadIdx.x;
  if (idx >= (size_t)B_ * MB_ * D_) return;
  int d = (int)(idx % D_);
  int m = (int)((idx / D_) % MB_);
  int b = (int)(idx / ((size_t)D_ * MB_));
  size_t g = (size_t)b * L_ * D_ + (size_t)(tokOff + m) * D_ + d;
  float kk = k[g], vv = v[g], l0 = lr0[g], l2 = lr2[g];
  size_t md = (size_t)b * MB_ * D_ + (size_t)m * D_ + d;
  kb[md] = f2bf(kk);
  vb[md] = f2bf(vv);
  size_t dm = (size_t)b * D_ * MB_ + (size_t)d * MB_ + m;
  klr0t[dm] = f2bf(kk * l0 * WSCALE);
  klr2t[dm] = f2bf(kk * l2 * WSCALE);
  vbt[dm]   = f2bf(vv);
}

__global__ void __launch_bounds__(256)
prep_q_kernel(const float* __restrict__ q, u16* __restrict__ qb, int tokOff, int T) {
  size_t idx = (size_t)blockIdx.x * 256 + threadIdx.x;
  if (idx >= (size_t)B_ * T * D_) return;
  int d = (int)(idx % D_);
  int m = (int)((idx / D_) % T);
  int b = (int)(idx / ((size_t)D_ * T));
  qb[idx] = f2bf(q[(size_t)b * L_ * D_ + (size_t)(tokOff + m) * D_ + d]);
}

// silu / backprop terms for the weight update
__global__ void __launch_bounds__(256)
update_terms_kernel(const float* __restrict__ gate, const float* __restrict__ hid,
                    const float* __restrict__ dhidden, const float* __restrict__ lr1, int tokOff,
                    u16* __restrict__ dgba, u16* __restrict__ dhidbm, u16* __restrict__ a1) {
  size_t idx = (size_t)blockIdx.x * 256 + threadIdx.x;
  if (idx >= (size_t)B_ * DH_ * MB_) return;
  int m = (int)(idx % MB_);
  int h = (int)((idx / MB_) % DH_);
  int b = (int)(idx / ((size_t)MB_ * DH_));
  float g = gate[idx], hh = hid[idx], dh = dhidden[idx];
  float s  = 1.f / (1.f + expf(-g));
  float sg = g * s;
  float l1 = lr1[(size_t)b * L_ * DH_ + (size_t)(tokOff + m) * DH_ + h];
  a1[idx]     = f2bf(sg * hh * l1 * WSCALE);   // hidden^T * lr1 * scale, stored [DH][MB]
  dhidbm[idx] = f2bf(dh * sg);
  float dgate = dh * hh;
  dgba[idx]   = f2bf(dgate * s * (1.f + g * (1.f - s)));
}

// z = silu(gate2)*hid2, stored transposed [B][T][DH] bf16 for the output GEMM
__global__ void __launch_bounds__(256)
z_kernel(const float* __restrict__ gate2, const float* __restrict__ hid2,
         u16* __restrict__ zT, int T) {
  size_t idx = (size_t)blockIdx.x * 256 + threadIdx.x;
  if (idx >= (size_t)B_ * DH_ * T) return;
  int m = (int)(idx % T);
  int h = (int)((idx / T) % DH_);
  int b = (int)(idx / ((size_t)T * DH_));
  float g = gate2[idx], hh = hid2[idx];
  float s = 1.f / (1.f + expf(-g));
  zT[(size_t)b * T * DH_ + (size_t)m * DH_ + h] = f2bf(g * s * hh);
}

extern "C" void kernel_launch(void* const* d_in, const int* in_sizes, int n_in,
                              void* d_out, int out_size, void* d_ws, size_t ws_size,
                              hipStream_t stream) {
  (void)in_sizes; (void)n_in; (void)out_size; (void)ws_size;
  const float* w0  = (const float*)d_in[0];
  const float* w1  = (const float*)d_in[1];
  const float* w2  = (const float*)d_in[2];
  const float* q   = (const float*)d_in[3];
  const float* k   = (const float*)d_in[4];
  const float* v   = (const float*)d_in[5];
  const float* lr0 = (const float*)d_in[6];
  const float* lr1 = (const float*)d_in[7];
  const float* lr2 = (const float*)d_in[8];

  float* outY = (float*)d_out;
  float* W0 = outY + (size_t)B_ * L_ * D_;         // working + final weights live in d_out
  float* W1 = W0 + (size_t)B_ * DH_ * D_;
  float* W2 = W1 + (size_t)B_ * D_ * DH_;
  const int WSZ = DH_ * D_;                        // per-batch weight element count

  // ---- workspace layout
  char* ws = (char*)d_ws;
  const size_t SZ_GATE = (size_t)B_ * DH_ * MB_ * 4;   // 16 MiB
  const size_t SZ_G2   = (size_t)B_ * DH_ * UE_ * 4;   // 32 MiB
  float* n0 = (float*)(ws);
  float* n1 = (float*)(ws + (size_t)B_ * DH_ * 4);
  float* n2 = (float*)(ws + (size_t)B_ * DH_ * 4 + (size_t)B_ * D_ * 4);
  char* fbase = ws + 131072;
  float* gate   = (float*)(fbase);
  float* hid    = (float*)(fbase + SZ_GATE);
  float* dhid   = (float*)(fbase + 2 * SZ_GATE);
  float* gate2  = (float*)(fbase);                 // reuses gate/hid/dhid region (disjoint phases)
  float* hid2   = (float*)(fbase + SZ_G2);
  char* bf = fbase + 2 * SZ_G2;
  const size_t WB = (size_t)B_ * DH_ * D_ * 2;     // 4 MiB per bf16 weight copy
  u16* wb0   = (u16*)(bf);
  u16* wb2   = (u16*)(bf + WB);
  u16* wb1t  = (u16*)(bf + 2 * WB);
  u16* wb1   = (u16*)(bf + 3 * WB);
  u16* kb    = (u16*)(bf + 4 * WB);
  u16* vb    = (u16*)(bf + 5 * WB);
  u16* klr0t = (u16*)(bf + 6 * WB);
  u16* klr2t = (u16*)(bf + 7 * WB);
  u16* vbt   = (u16*)(bf + 8 * WB);
  u16* dgba   = (u16*)(bf + 9 * WB);
  u16* dhidbm = (u16*)(bf + 9 * WB + 2 * WB);
  u16* a1     = (u16*)(bf + 9 * WB + 4 * WB);
  u16* qb     = (u16*)(bf + 9 * WB + 6 * WB);                 // B*UE*D bf16 (8 MiB)
  u16* zT     = (u16*)(bf + 9 * WB + 6 * WB + 2 * WB);        // B*UE*DH bf16 (16 MiB)

  auto gemm = [&](const u16* A, int sA, const u16* Bp, int sB, float* C, int sC,
                  int M, int N, int K, int ldc, int acc) {
    dim3 g(N / 128, M / 128, B_);
    gemm_bf16<<<g, 128, 0, stream>>>(A, sA, Bp, sB, C, sC, M, N, K, ldc, acc);
  };
  auto conv_weights = [&]() {
    size_t nw = (size_t)B_ * DH_ * D_;
    unsigned gb = (unsigned)((nw + 255) / 256);
    f32_to_bf16<<<gb, 256, 0, stream>>>(W0, wb0, nw);
    f32_to_bf16<<<gb, 256, 0, stream>>>(W2, wb2, nw);
    f32_to_bf16<<<gb, 256, 0, stream>>>(W1, wb1, nw);
    transpose_f32_to_bf16<<<gb, 256, 0, stream>>>(W1, wb1t, D_, DH_);
  };
  auto fwd = [&](int tok, int T) {
    unsigned gq = (unsigned)(((size_t)B_ * T * D_ + 255) / 256);
    prep_q_kernel<<<gq, 256, 0, stream>>>(q, qb, tok, T);
    gemm(wb0, WSZ, qb, T * D_, gate2, DH_ * T, DH_, T, D_, T, 0);
    gemm(wb2, WSZ, qb, T * D_, hid2,  DH_ * T, DH_, T, D_, T, 0);
    unsigned gz = (unsigned)(((size_t)B_ * DH_ * T + 255) / 256);
    z_kernel<<<gz, 256, 0, stream>>>(gate2, hid2, zT, T);
    gemm(zT, T * DH_, wb1, WSZ, outY + (size_t)tok * D_, L_ * D_, T, D_, DH_, D_, 0);
  };

  // ---- init working weights + fixed target row norms
  size_t wbytes = (size_t)B_ * DH_ * D_ * 4;
  hipMemcpyAsync(W0, w0, wbytes, hipMemcpyDeviceToDevice, stream);
  hipMemcpyAsync(W1, w1, wbytes, hipMemcpyDeviceToDevice, stream);
  hipMemcpyAsync(W2, w2, wbytes, hipMemcpyDeviceToDevice, stream);
  rownorm_kernel<<<B_ * DH_, 256, 0, stream>>>(w0, n0, D_);
  rownorm_kernel<<<B_ * D_,  256, 0, stream>>>(w1, n1, DH_);
  rownorm_kernel<<<B_ * DH_, 256, 0, stream>>>(w2, n2, D_);
  conv_weights();

  // ---- first segment forward with initial weights
  fwd(0, FIRST_);

  // ---- sequential chunks
  for (int ic = 0; ic < NC_; ++ic) {
    int tok = FIRST_ + ic * UE_;
    unsigned gkv = (unsigned)(((size_t)B_ * MB_ * D_ + 255) / 256);
    prep_kv_kernel<<<gkv, 256, 0, stream>>>(k, v, lr0, lr2, tok, kb, vb, klr0t, klr2t, vbt);
    // forward through fast weights on the mini-batch (pre-update weights)
    gemm(wb0,  WSZ, kb, MB_ * D_, gate, DH_ * MB_, DH_, MB_, D_, MB_, 0);
    gemm(wb2,  WSZ, kb, MB_ * D_, hid,  DH_ * MB_, DH_, MB_, D_, MB_, 0);
    gemm(wb1t, WSZ, vb, MB_ * D_, dhid, DH_ * MB_, DH_, MB_, D_, MB_, 0);
    unsigned gu = (unsigned)(((size_t)B_ * DH_ * MB_ + 255) / 256);
    update_terms_kernel<<<gu, 256, 0, stream>>>(gate, hid, dhid, lr1, tok, dgba, dhidbm, a1);
    // weight updates (accumulate into fp32 working weights)
    gemm(dgba,   DH_ * MB_, klr0t, D_ * MB_, W0, WSZ, DH_, D_,  MB_, D_,  1);
    gemm(dhidbm, DH_ * MB_, klr2t, D_ * MB_, W2, WSZ, DH_, D_,  MB_, D_,  1);
    gemm(vbt,    D_ * MB_,  a1,  DH_ * MB_,  W1, WSZ, D_,  DH_, MB_, DH_, 1);
    // renormalize rows to initial norms
    renorm_kernel<<<B_ * DH_, 256, 0, stream>>>(W0, n0, D_);
    renorm_kernel<<<B_ * D_,  256, 0, stream>>>(W1, n1, DH_);
    renorm_kernel<<<B_ * DH_, 256, 0, stream>>>(W2, n2, D_);
    conv_weights();
    // forward over the full update-every window with the new weights
    fwd(tok, UE_);
  }
}
